// Closest_embedding_layer_21492016349994
// MI455X (gfx1250) — matrix operand order repaired and checked by reading, not compiled, not run on hardware
//
#include <hip/hip_runtime.h>

// Problem constants (from reference): B=2048, U=64, K=1024, D=256
#define NROWS   131072       // B*U flattened query rows
#define DDIM    256
#define KCODES  1024
#define TILES   (KCODES / 16)
#define ROWS_PER_WAVE 32     // two 16-row WMMA groups sharing each B fragment
#define MBLOCK  256          // 8 waves x 32 rows
#define ZSTRIDE 264          // 256 + 8 ushort pad -> conflict-free ds_read_b128
#define ESTRIDE 264

typedef __attribute__((ext_vector_type(16))) __bf16 v16bf;
typedef __attribute__((ext_vector_type(8)))  float  v8f;

static __device__ __forceinline__ unsigned short f2bf(float f) {
  union { float f; unsigned u; } v; v.f = f;
  unsigned r = v.u + 0x7FFFu + ((v.u >> 16) & 1u);   // round-to-nearest-even
  return (unsigned short)(r >> 16);
}

// ---------------------------------------------------------------------------
// Kernel 1: codebook prep. E (f32 [K,D]) -> Ebf (bf16 [K,D]) and esq[k]=||E_k||^2.
// ---------------------------------------------------------------------------
__global__ void __launch_bounds__(256) prep_codebook(const float* __restrict__ E,
                                                     unsigned short* __restrict__ Ebf,
                                                     float* __restrict__ esq) {
  const int wv   = threadIdx.x >> 5;
  const int lane = threadIdx.x & 31;
  const int row  = blockIdx.x * 8 + wv;
  const float* src = E + row * DDIM + lane * 8;
  float4 f0 = ((const float4*)src)[0];
  float4 f1 = ((const float4*)src)[1];
  uint4 pk;
  pk.x = (unsigned)f2bf(f0.x) | ((unsigned)f2bf(f0.y) << 16);
  pk.y = (unsigned)f2bf(f0.z) | ((unsigned)f2bf(f0.w) << 16);
  pk.z = (unsigned)f2bf(f1.x) | ((unsigned)f2bf(f1.y) << 16);
  pk.w = (unsigned)f2bf(f1.z) | ((unsigned)f2bf(f1.w) << 16);
  *(uint4*)(Ebf + row * DDIM + lane * 8) = pk;
  float s = f0.x*f0.x + f0.y*f0.y + f0.z*f0.z + f0.w*f0.w
          + f1.x*f1.x + f1.y*f1.y + f1.z*f1.z + f1.w*f1.w;
  #pragma unroll
  for (int m = 16; m >= 1; m >>= 1) s += __shfl_xor(s, m, 32);
  if (lane == 0) esq[row] = s;
}

// ---------------------------------------------------------------------------
// Kernel 2: 256-row tile per block. score = z.E^T via v_wmma_f32_16x16x32_bf16
// with LDS-staged, double-buffered E tiles shared by all 8 waves; each B
// fragment feeds two independent 16-row accumulation chains. Running argmin of
// (esq[k] - 2*score), then coalesced gather of E[k*] (f32).
// ---------------------------------------------------------------------------
__global__ void __launch_bounds__(256) vq_argmin_gather(
    const float* __restrict__ z,
    const float* __restrict__ E,
    const unsigned short* __restrict__ Ebf,
    const float* __restrict__ esq,
    float* __restrict__ out) {

  __shared__ unsigned short zt[MBLOCK * ZSTRIDE];       // 132 KB bf16 z tile
  __shared__ unsigned short et[2][16 * ESTRIDE];        // 2 x 8.25 KB E tile
  __shared__ int bestk[MBLOCK];

  const int  tid      = threadIdx.x;
  const long row_base = (long)blockIdx.x * MBLOCK;

  // ---- stage z tile (f32 global, coalesced float4) -> bf16 LDS -------------
  for (int idx = tid; idx < MBLOCK * (DDIM / 4); idx += 256) {
    const int r  = idx >> 6;      // 64 float4 per row
    const int c4 = idx & 63;
    float4 f = ((const float4*)(z + (row_base + r) * DDIM))[c4];
    uint2 pk;
    pk.x = (unsigned)f2bf(f.x) | ((unsigned)f2bf(f.y) << 16);
    pk.y = (unsigned)f2bf(f.z) | ((unsigned)f2bf(f.w) << 16);
    *(uint2*)(zt + r * ZSTRIDE + c4 * 4) = pk;
  }

  // ---- stage E tile 0 (bf16 global -> LDS, 32B per thread) -----------------
  const int srow = tid >> 4;            // 16 rows
  const int scol = (tid & 15) * 16;     // 16 ushorts (32B) per thread
  {
    const uint4* sp = (const uint4*)(Ebf + (size_t)srow * DDIM + scol);
    uint4* dp = (uint4*)(et[0] + srow * ESTRIDE + scol);
    dp[0] = sp[0]; dp[1] = sp[1];
  }
  __syncthreads();

  const int wv    = tid >> 5;
  const int lane  = tid & 31;
  const int h     = lane >> 4;          // lane half (ISA A/B frag layout)
  const int mycol = lane & 15;          // C/D column = code within tile

  // ---- all 16 A-fragments (2 row groups x full D=256) resident in regs -----
  // A 16x32 bf16 layout: lane half h holds K = {h*8..h*8+7} U {h*8+16..h*8+23}
  union AF { v16bf v; uint4 q[2]; };
  AF a[2][8];
  #pragma unroll
  for (int g = 0; g < 2; ++g) {
    const unsigned short* zrow = zt + (wv * 32 + g * 16 + mycol) * ZSTRIDE;
    #pragma unroll
    for (int kk = 0; kk < 8; ++kk) {
      const int kb = kk * 32 + h * 8;
      a[g][kk].q[0] = *(const uint4*)(zrow + kb);        // K = kb .. kb+7
      a[g][kk].q[1] = *(const uint4*)(zrow + kb + 16);   // K = kb+16 .. kb+23
    }
  }

  float bd[2][8];
  int   bc[2][8];
  #pragma unroll
  for (int g = 0; g < 2; ++g)
    #pragma unroll
    for (int i = 0; i < 8; ++i) { bd[g][i] = 3.4e38f; bc[g][i] = 0; }

  // ---- sweep 1024 codes in tiles of 16, double-buffered through LDS --------
  for (int t = 0; t < TILES; ++t) {
    // issue next tile's global loads early (overlap with WMMA below)
    uint4 s0, s1;
    if (t + 1 < TILES) {
      const uint4* sp =
          (const uint4*)(Ebf + (size_t)((t + 1) * 16 + srow) * DDIM + scol);
      s0 = sp[0]; s1 = sp[1];
    }

    // B 32x16 bf16 layout: lane l holds E row (lane%16), K = kk*32 + {h*8..h*8+7}
    // U {h*8+16..h*8+23} -> two ds_read_b128 per fragment, shared by both
    // row-group WMMA chains.
    const unsigned short* eb = et[t & 1] + mycol * ESTRIDE + h * 8;
    v8f acc0 = {}, acc1 = {};
    #pragma unroll
    for (int kk = 0; kk < 8; ++kk) {
      union BF { v16bf v; uint4 q[2]; } b;
      b.q[0] = *(const uint4*)(eb + kk * 32);
      b.q[1] = *(const uint4*)(eb + kk * 32 + 16);
      acc0 = __builtin_amdgcn_wmma_f32_16x16x32_bf16(
          false, a[0][kk].v, false, b.v, (short)0, acc0, false, false);
      acc1 = __builtin_amdgcn_wmma_f32_16x16x32_bf16(
          false, a[1][kk].v, false, b.v, (short)0, acc1, false, false);
    }

    // complete the next-tile staging store
    if (t + 1 < TILES) {
      uint4* dp = (uint4*)(et[(t + 1) & 1] + srow * ESTRIDE + scol);
      dp[0] = s0; dp[1] = s1;
    }

    const int  code = t * 16 + mycol;
    const float es  = esq[code];
    #pragma unroll
    for (int i = 0; i < 8; ++i) {
      const float d0 = fmaf(-2.0f, acc0[i], es);   // ||z||^2 rank-invariant
      const float d1 = fmaf(-2.0f, acc1[i], es);
      if (d0 < bd[0][i]) { bd[0][i] = d0; bc[0][i] = code; }
      if (d1 < bd[1][i]) { bd[1][i] = d1; bc[1][i] = code; }
    }
    __syncthreads();   // buffer flip
  }

  // ---- argmin across the 16 lanes of each half (C row = i + 8*h) -----------
  #pragma unroll
  for (int g = 0; g < 2; ++g) {
    #pragma unroll
    for (int i = 0; i < 8; ++i) {
      float d = bd[g][i];
      int   c = bc[g][i];
      #pragma unroll
      for (int m = 8; m >= 1; m >>= 1) {
        const float od = __shfl_xor(d, m, 32);
        const int   oc = __shfl_xor(c, m, 32);
        if (od < d || (od == d && oc < c)) { d = od; c = oc; }  // lowest-k ties
      }
      bc[g][i] = c;
    }
  }
  if ((lane & 15) == 0) {
    #pragma unroll
    for (int g = 0; g < 2; ++g)
      #pragma unroll
      for (int i = 0; i < 8; ++i)
        bestk[wv * 32 + g * 16 + h * 8 + i] = bc[g][i];
  }
  __syncthreads();

  // ---- gather winning code rows (original f32 E), coalesced 1KB/row --------
  for (int r = wv; r < MBLOCK; r += 8) {
    const int k = bestk[r];
    const float4* src = (const float4*)(E + (size_t)k * DDIM);
    float4* dst = (float4*)(out + (row_base + r) * DDIM);
    dst[lane]      = src[lane];
    dst[lane + 32] = src[lane + 32];
  }
}

// ---------------------------------------------------------------------------
extern "C" void kernel_launch(void* const* d_in, const int* in_sizes, int n_in,
                              void* d_out, int out_size, void* d_ws, size_t ws_size,
                              hipStream_t stream) {
  (void)in_sizes; (void)n_in; (void)out_size; (void)ws_size;
  const float* z = (const float*)d_in[0];   // [B,U,D] f32
  const float* E = (const float*)d_in[1];   // [K,D]   f32

  unsigned short* Ebf = (unsigned short*)d_ws;                       // 512 KB
  float* esq = (float*)((char*)d_ws + (size_t)KCODES * DDIM * 2);    // 4 KB
  float* out = (float*)d_out;

  prep_codebook<<<KCODES / 8, 256, 0, stream>>>(E, Ebf, esq);
  vq_argmin_gather<<<NROWS / MBLOCK, 256, 0, stream>>>(z, E, Ebf, esq, out);
}
